// RSPLRNN_70042326663991
// MI455X (gfx1250) — matrix operand order repaired and checked by reading, not compiled, not run on hardware
//
#include <hip/hip_runtime.h>
#include <hip/hip_bf16.h>
#include <stdint.h>

// ---- CDNA5 WMMA types -------------------------------------------------------
typedef __attribute__((ext_vector_type(16))) __bf16        v16bf;
typedef __attribute__((ext_vector_type(8)))  float         v8f;
typedef __attribute__((ext_vector_type(4)))  unsigned int  v4u;

union Frag { v4u q[2]; v16bf v; };   // 8 dwords <-> 16 bf16 (VGPR0..7)

#define HID   1024
#define INSZ  512
#define BATCH 64
#define TLEN  512
#define KC    512            // K staging chunk in LDS
#define KPAD  (KC + 8)       // row stride 520 bf16 = 260 dwords (== 4 mod 64: bank-conflict free)

__device__ __forceinline__ unsigned short f2bf(float f) {
  unsigned u = __float_as_uint(f);
  u += 0x7FFFu + ((u >> 16) & 1u);           // round-to-nearest-even
  return (unsigned short)(u >> 16);
}

// -----------------------------------------------------------------------------
// Pack fp32 weight [K x N] (row-major, N contiguous) into WMMA-B fragment order
// bf16. Per (nt,kt) 16x32 N-x-K tile: 256 dwords, dword index = lane*8 + j.
// B layout (contiguous k order): lanes 0-15 -> N=nt*16+lane, K = kt*32 + 2j, 2j+1
//                        lanes 16-31 -> same N set,  K = kt*32 + 16 + 2j, +1
// -----------------------------------------------------------------------------
__global__ void pack_b(const float* __restrict__ W, unsigned int* __restrict__ dst,
                       int K, int N) {
  const int KT = K >> 5;
  const long long total = (long long)(N >> 4) * KT * 256;  // dwords
  long long t = (long long)blockIdx.x * blockDim.x + threadIdx.x;
  if (t >= total) return;
  const int within = (int)(t & 255);
  const long long blk = t >> 8;
  const int kt  = (int)(blk % KT);
  const int nt  = (int)(blk / KT);
  const int lane = within >> 3;
  const int j    = within & 7;
  const int n  = nt * 16 + (lane & 15);
  const int k0 = kt * 32 + ((lane < 16) ? 0 : 16) + 2 * j;
  const unsigned short b0 = f2bf(W[(size_t)k0 * N + n]);
  const unsigned short b1 = f2bf(W[(size_t)(k0 + 1) * N + n]);
  dst[t] = ((unsigned int)b1 << 16) | (unsigned int)b0;
}

// -----------------------------------------------------------------------------
// Persistent RNN kernel: 4 workgroups x 1024 threads (32 waves).
// WG b owns batch rows [16b, 16b+16); wave w owns output cols [32w, 32w+32).
// All recurrence state is per-WG-private -> only __syncthreads() needed.
// -----------------------------------------------------------------------------
__global__ void __launch_bounds__(1024)
rspl_rnn(const float* __restrict__ X,       // [T, B, INSZ]
         const float* __restrict__ hinit,   // [2, B, HID]
         const unsigned char* __restrict__ Spk0,
         const unsigned char* __restrict__ Wpk0,
         const unsigned char* __restrict__ Spk1,
         const unsigned char* __restrict__ Wpk1,
         float* __restrict__ h0,            // [B, HID] ws
         float* __restrict__ h1,            // [B, HID] ws
         float* __restrict__ out)           // [T,B,HID] ++ [2,B,HID]
{
  __shared__ unsigned short sA[16 * KPAD];  // bf16 p-tile chunk
  __shared__ float sRow[16];                // per-row sum of u^2

  const int tid  = threadIdx.x;
  const int lane = tid & 31;
  const int wave = tid >> 5;
  const int half = lane >> 4;        // 0: lanes 0-15, 1: lanes 16-31
  const int l16  = lane & 15;
  const int r0   = blockIdx.x * 16;  // first batch row owned by this WG
  const int nb   = wave * 32;        // first output col owned by this wave

  // init recurrent state from hinit (deterministic every call)
  for (int i = tid; i < 16 * HID; i += 1024) {
    const int row = i >> 10, col = i & (HID - 1);
    h0[(size_t)(r0 + row) * HID + col] = hinit[(size_t)(0 * BATCH + r0 + row) * HID + col];
    h1[(size_t)(r0 + row) * HID + col] = hinit[(size_t)(1 * BATCH + r0 + row) * HID + col];
  }
  __syncthreads();

  for (int t = 0; t < TLEN; ++t) {
    for (int layer = 0; layer < 2; ++layer) {
      const int K  = layer ? (HID + HID) : (HID + INSZ);
      const int KT = K >> 5;
      float* hcur = layer ? h1 : h0;
      const unsigned char* Spk = layer ? Spk1 : Spk0;
      const unsigned char* Wpk = layer ? Wpk1 : Wpk0;

      v8f accS0 = {}, accS1 = {}, accW0 = {}, accW1 = {};

      for (int kc = 0; kc < K; kc += KC) {
        __syncthreads();   // prior pA / sRow consumers done
        if (kc == 0 && tid < 16) sRow[tid] = 0.f;
        // stage bf16 p = [h_prev | x] chunk into LDS (coalesced reads)
        for (int i = tid; i < 16 * KC; i += 1024) {
          const int row = i >> 9;
          const int e   = i & (KC - 1);
          const int eg  = kc + e;
          float v;
          if (eg < HID)          v = hcur[(size_t)(r0 + row) * HID + eg];
          else if (layer == 0)   v = X[((size_t)t * BATCH + (r0 + row)) * INSZ + (eg - HID)];
          else                   v = h0[(size_t)(r0 + row) * HID + (eg - HID)];
          sA[row * KPAD + e] = f2bf(v);
        }
        __syncthreads();

        // A frag addressing (contiguous k order, as pack_b)
        const int abase = l16 * KPAD + (half ? 16 : 0);
        for (int kk = 0; kk < KC; kk += 32) {
          Frag a;
          a.q[0] = *(const v4u*)&sA[abase + kk];        // K +0..7  (or +16..23)
          a.q[1] = *(const v4u*)&sA[abase + kk + 8];    // K +8..15 (or +24..31)

          const int kt = (kc + kk) >> 5;
          const size_t blk0 = ((size_t)(wave * 2 + 0) * KT + kt) * 1024 + (size_t)lane * 32;
          const size_t blk1 = ((size_t)(wave * 2 + 1) * KT + kt) * 1024 + (size_t)lane * 32;
          __builtin_prefetch((const void*)(Spk + blk0 + 2048), 0, 3);
          __builtin_prefetch((const void*)(Wpk + blk0 + 2048), 0, 3);

          Frag bS0, bS1, bW0, bW1;
          bS0.q[0] = *(const v4u*)(Spk + blk0);  bS0.q[1] = *(const v4u*)(Spk + blk0 + 16);
          bS1.q[0] = *(const v4u*)(Spk + blk1);  bS1.q[1] = *(const v4u*)(Spk + blk1 + 16);
          bW0.q[0] = *(const v4u*)(Wpk + blk0);  bW0.q[1] = *(const v4u*)(Wpk + blk0 + 16);
          bW1.q[0] = *(const v4u*)(Wpk + blk1);  bW1.q[1] = *(const v4u*)(Wpk + blk1 + 16);

          accS0 = __builtin_amdgcn_wmma_f32_16x16x32_bf16(false, a.v, false, bS0.v, (short)0, accS0, false, false);
          accS1 = __builtin_amdgcn_wmma_f32_16x16x32_bf16(false, a.v, false, bS1.v, (short)0, accS1, false, false);
          accW0 = __builtin_amdgcn_wmma_f32_16x16x32_bf16(false, a.v, false, bW0.v, (short)0, accW0, false, false);
          accW1 = __builtin_amdgcn_wmma_f32_16x16x32_bf16(false, a.v, false, bW1.v, (short)0, accW1, false, false);
        }
      }

      // ---- epilogue: z = sigmoid(pS), u = (1-z)h + z(pW), per-row ||u|| -----
      float u0[8], u1[8], part[8];
      #pragma unroll
      for (int j = 0; j < 8; ++j) {
        const int m = j + half * 8;                      // local row (C/D layout)
        const size_t ro = (size_t)(r0 + m) * HID;
        const float hp0 = hcur[ro + nb + l16];
        const float hp1 = hcur[ro + nb + 16 + l16];
        const float z0 = 1.f / (1.f + __expf(-accS0[j]));
        const float z1 = 1.f / (1.f + __expf(-accS1[j]));
        u0[j] = (1.f - z0) * hp0 + z0 * accW0[j];
        u1[j] = (1.f - z1) * hp1 + z1 * accW1[j];
        part[j] = u0[j] * u0[j] + u1[j] * u1[j];
      }
      // reduce over the 16 lanes sharing the same row set
      #pragma unroll
      for (int mask = 1; mask <= 8; mask <<= 1) {
        #pragma unroll
        for (int j = 0; j < 8; ++j)
          part[j] += __shfl_xor(part[j], mask, 32);
      }
      if (l16 == 0) {
        #pragma unroll
        for (int j = 0; j < 8; ++j)
          atomicAdd(&sRow[j + half * 8], part[j]);       // ds_add_f32
      }
      __syncthreads();

      #pragma unroll
      for (int j = 0; j < 8; ++j) {
        const int m = j + half * 8;
        const float inv = 1.f / (sqrtf(sRow[m]) + 1e-8f);
        const float v0 = u0[j] * inv;
        const float v1 = u1[j] * inv;
        const size_t ro = (size_t)(r0 + m) * HID;
        hcur[ro + nb + l16]      = v0;
        hcur[ro + nb + 16 + l16] = v1;
        if (layer == 1) {
          const size_t o = ((size_t)t * BATCH + (r0 + m)) * HID;
          out[o + nb + l16]      = v0;
          out[o + nb + 16 + l16] = v1;
        }
        if (t == TLEN - 1) {
          const size_t o = (size_t)TLEN * BATCH * HID
                         + ((size_t)layer * BATCH + (r0 + m)) * HID;
          out[o + nb + l16]      = v0;
          out[o + nb + 16 + l16] = v1;
        }
      }
    }
  }
}

// -----------------------------------------------------------------------------
extern "C" void kernel_launch(void* const* d_in, const int* in_sizes, int n_in,
                              void* d_out, int out_size, void* d_ws, size_t ws_size,
                              hipStream_t stream) {
  const float* X  = (const float*)d_in[0];
  const float* h0i= (const float*)d_in[1];
  const float* S0 = (const float*)d_in[2];
  const float* W0 = (const float*)d_in[3];
  const float* S1 = (const float*)d_in[4];
  const float* W1 = (const float*)d_in[5];
  float* out = (float*)d_out;

  unsigned char* ws = (unsigned char*)d_ws;
  // ws layout: Spk0(3MB) Wpk0(3MB) Spk1(4MB) Wpk1(4MB) h0(256KB) h1(256KB)
  unsigned char* Spk0 = ws;
  unsigned char* Wpk0 = ws + (3ull << 20);
  unsigned char* Spk1 = ws + (6ull << 20);
  unsigned char* Wpk1 = ws + (10ull << 20);
  float* h0b = (float*)(ws + (14ull << 20));
  float* h1b = (float*)(ws + (14ull << 20) + (256ull << 10));

  const int N = 1024, K0 = 1536, K1 = 2048, blk = 256;
  const long long tot0 = (long long)N * K0 / 2;   // packed dwords
  const long long tot1 = (long long)N * K1 / 2;
  pack_b<<<(unsigned)((tot0 + blk - 1) / blk), blk, 0, stream>>>(S0, (unsigned int*)Spk0, K0, N);
  pack_b<<<(unsigned)((tot0 + blk - 1) / blk), blk, 0, stream>>>(W0, (unsigned int*)Wpk0, K0, N);
  pack_b<<<(unsigned)((tot1 + blk - 1) / blk), blk, 0, stream>>>(S1, (unsigned int*)Spk1, K1, N);
  pack_b<<<(unsigned)((tot1 + blk - 1) / blk), blk, 0, stream>>>(W1, (unsigned int*)Wpk1, K1, N);

  rspl_rnn<<<4, 1024, 0, stream>>>(X, h0i, Spk0, Wpk0, Spk1, Wpk1, h0b, h1b, out);
}
